// LightningAttention_62758062129468
// MI455X (gfx1250) — matrix operand order repaired and checked by reading, not compiled, not run on hardware
//
#include <hip/hip_runtime.h>

// Lightning attention, chunked scan, f16 WMMA with fp32 accumulation.
// Shapes hardcoded from the reference: B=2,H=16,N=4096,D=128,BLOCK=128.

typedef __attribute__((ext_vector_type(16))) _Float16 v16h;
typedef __attribute__((ext_vector_type(8)))  float    v8f;

#define BB   2
#define HH   16
#define NN   4096
#define DD   128
#define BLK  128
#define EW   32            // output-column slice per workgroup
#define NSPL (DD / EW)     // 4 splits -> 128 workgroups
#define NCH  (NN / BLK)    // 32 chunks
#define NTHR 256
#define NWAVE 8

// ---- WMMA fragment loaders (LDS, row stride 128 f16) --------------------
// A fragment (16x32 f16, MxK): lane l(0..15)=row, lane half h selects K-halves.
//   elems 0..7  -> K = k0 + h*8 + j
//   elems 8..15 -> K = k0 + 16 + h*8 + j
__device__ __forceinline__ v16h frag_a(const _Float16* X, int m0, int k0, int lane) {
    int l = lane & 15, h = lane >> 4;
    const _Float16* p = X + (m0 + l) * 128 + k0 + h * 8;
    v16h a;
#pragma unroll
    for (int j = 0; j < 8; ++j) { a[j] = p[j]; a[8 + j] = p[16 + j]; }
    return a;
}
// B fragment (32x16 f16, KxN) loaded from a K-contiguous ("transposed") buffer:
//   Y is stored as Y[n][k]; b[j] = B[k0 + (lane>>4)*8 + j, n], b[8+j] = B[k0 + 16 + (lane>>4)*8 + j, n], n = n0 + (lane&15)
__device__ __forceinline__ v16h frag_b(const _Float16* Y, int n0, int k0, int lane) {
    int l = lane & 15, h = lane >> 4;
    const _Float16* p = Y + (n0 + l) * 128 + k0 + h * 8;
    v16h b;
#pragma unroll
    for (int j = 0; j < 8; ++j) { b[j] = p[j]; b[8 + j] = p[16 + j]; }
    return b;
}
__device__ __forceinline__ v8f wmma16(v16h a, v16h b, v8f c) {
    return __builtin_amdgcn_wmma_f32_16x16x32_f16(false, a, false, b, (short)0, c,
                                                  false, false);
}

__global__ void lightning_attn_kernel(const float* __restrict__ q,
                                      const float* __restrict__ k,
                                      const float* __restrict__ v,
                                      const float* __restrict__ s,
                                      float* __restrict__ out) {
    const int tid  = threadIdx.x;
    const int lane = tid & 31;
    const int w    = tid >> 5;           // wave id 0..7 (uniform per wave)
    const int l16  = lane & 15;
    const int hh   = lane >> 4;

    const int bh  = blockIdx.x / NSPL;   // (b*H + h)
    const int esl = blockIdx.x % NSPL;   // which 32-column slice
    const int h   = bh % HH;
    const float slope = s[h];
    const float bdec  = __expf(-slope * (float)BLK);   // lambda^BLOCK

    // lam[r] = lambda^r, r=0..7 (loop-invariant per-thread constants)
    float lam[8];
#pragma unroll
    for (int r = 0; r < 8; ++r) lam[r] = __expf(-slope * (float)r);

    // ---- LDS carve-up (~178 KiB dynamic) ----
    extern __shared__ __align__(16) char smem_raw[];
    float*    qdec = (float*)smem_raw;            // 128
    float*    kdec = qdec + 128;                  // 128
    float*    dtab = kdec + 128;                  // 256: exp(-slope*d) / 0 for d>=128
    _Float16* qh   = (_Float16*)(dtab + 256);     // 128x128 raw q
    _Float16* qlh  = qh  + 128 * 128;             // 128x128 q * q_decay[t]
    _Float16* kh   = qlh + 128 * 128;             // 128x128 raw k  (rows=s, cols=d)
    _Float16* kmT  = kh  + 128 * 128;             // 128x128 (k*k_decay)^T (rows=d, cols=s)
    _Float16* vT   = kmT + 128 * 128;             // 32x128  v slice transposed (rows=e, cols=s)
    _Float16* ShT  = vT  + EW * 128;              // 32x128  f16 S^T (rows=e, cols=d)
    _Float16* sch  = ShT + EW * 128;              // 128x128 decayed scores f16

    // one-time init: decay tables + zero scores buffer (upper-tri tiles stay 0)
    if (tid < 128) {
        qdec[tid] = __expf(-slope * (float)tid);
        kdec[tid] = __expf(-slope * (float)(BLK - tid));
    }
    // dtab[(t-s)&255]: exp(-slope*(t-s)) for t>=s, 0 for t<s (wrapped negatives)
    dtab[tid] = (tid < 128) ? __expf(-slope * (float)tid) : 0.0f;
    for (int i = tid; i < 128 * 128; i += NTHR) sch[i] = (_Float16)0.0f;

    // persistent fp32 state tiles: wave w owns d-rows [16w,16w+16), e-cols [0,32)
    v8f Sacc[2];
    Sacc[0] = {}; Sacc[1] = {};

    const size_t qkbase0 = (size_t)bh * NN * DD;
    const size_t obase0  = (size_t)bh * NN * DD;

    __syncthreads();

#pragma unroll 1
    for (int c0 = 0; c0 < NCH; ++c0) {
        const size_t cb = qkbase0 + (size_t)c0 * BLK * DD;

        // ---- 1) publish f16 S^T for this chunk's o_inter ----
#pragma unroll
        for (int eB = 0; eB < 2; ++eB)
#pragma unroll
            for (int r = 0; r < 8; ++r) {
                int d = w * 16 + hh * 8 + r;
                int e = eB * 16 + l16;
                ShT[e * 128 + d] = (_Float16)Sacc[eB][r];
            }

        // ---- prefetch next chunk's q/k (global_prefetch_b8) ----
        if (c0 + 1 < NCH) {
            const float* qn = q + cb + BLK * DD;
            const float* kn = k + cb + BLK * DD;
#pragma unroll
            for (int i = 0; i < 64; i += 16) {
                __builtin_prefetch(qn + i * NTHR + tid, 0, 3);
                __builtin_prefetch(kn + i * NTHR + tid, 0, 3);
            }
        }

        // ---- cooperative global -> LDS staging (f16 convert + decay) ----
#pragma unroll 4
        for (int i = 0; i < 64; ++i) {
            int idx = i * NTHR + tid;          // 0..16383
            int t = idx >> 7, d = idx & 127;
            float qv = q[cb + idx];
            qh[t * 128 + d]  = (_Float16)qv;
            qlh[t * 128 + d] = (_Float16)(qv * qdec[t]);
            float kv = k[cb + idx];
            kh[t * 128 + d]  = (_Float16)kv;
            kmT[d * 128 + t] = (_Float16)(kv * kdec[t]);
        }
#pragma unroll 4
        for (int i = 0; i < 16; ++i) {
            int idx = i * NTHR + tid;          // 0..4095
            int ss = idx >> 5, e = idx & 31;
            vT[e * 128 + ss] = (_Float16)v[cb + (size_t)ss * DD + esl * EW + e];
        }
        __syncthreads();

        // ---- 2) scores = q k^T, lower-triangular tiles, decay -> LDS f16 ----
        {
            int idx = 0;
            for (int tB = 0; tB < 8; ++tB)
                for (int sB = 0; sB <= tB; ++sB, ++idx) {
                    if ((idx & 7) != w) continue;            // wave-uniform
                    v8f acc = {};
#pragma unroll
                    for (int kk = 0; kk < 4; ++kk) {
                        v16h a = frag_a(qh, tB * 16, kk * 32, lane);
                        v16h b = frag_b(kh, sB * 16, kk * 32, lane);  // B = k^T
                        acc = wmma16(a, b, acc);
                    }
                    _Float16* srow = sch + (tB * 16 + hh * 8) * 128 + sB * 16 + l16;
                    if (tB == sB) {
                        // diagonal tile: sign of (t-s) varies -> branchless LUT
#pragma unroll
                        for (int r = 0; r < 8; ++r) {
                            int diff = hh * 8 + r - l16;
                            float dm = dtab[diff & 255];
                            srow[r * 128] = (_Float16)(acc[r] * dm);
                        }
                    } else {
                        // off-diagonal: t-s >= 1 everywhere ->
                        // dm(r) = dm0 * lambda^r, one LDS lookup per tile
                        float dm0 = dtab[(tB - sB) * 16 + hh * 8 - l16];
#pragma unroll
                        for (int r = 0; r < 8; ++r)
                            srow[r * 128] = (_Float16)(acc[r] * lam[r] * dm0);
                    }
                }
        }
        __syncthreads();

        // ---- 3+4 fused per eB (unroll 1 -> low register pressure, no spills):
        //      o  = scores @ v + (q*qdec) @ S_old
        //      S  = lambda^B * S + (k*kdec)^T @ v     (vT B-fragments reused)
#pragma unroll 1
        for (int eB = 0; eB < 2; ++eB) {
            v8f oacc = {};
#pragma unroll
            for (int r = 0; r < 8; ++r) Sacc[eB][r] *= bdec;
#pragma unroll
            for (int kk = 0; kk < 4; ++kk) {
                v16h bv = frag_b(vT, eB * 16, kk * 32, lane);   // shared B frag
                v16h a1 = frag_a(sch, w * 16, kk * 32, lane);
                oacc = wmma16(a1, bv, oacc);
                v16h a2 = frag_a(kmT, w * 16, kk * 32, lane);   // A = (k*mu)^T
                Sacc[eB] = wmma16(a2, bv, Sacc[eB]);
            }
#pragma unroll
            for (int kk = 0; kk < 4; ++kk) {
                v16h a = frag_a(qlh, w * 16, kk * 32, lane);
                v16h b = frag_b(ShT, eB * 16, kk * 32, lane);
                oacc = wmma16(a, b, oacc);
            }
            int col = esl * EW + eB * 16 + l16;
#pragma unroll
            for (int r = 0; r < 8; ++r) {
                int t = w * 16 + hh * 8 + r;
                out[obase0 + (size_t)(c0 * BLK + t) * DD + col] = oacc[r];
            }
        }
        __syncthreads();   // protect LDS (incl. ShT) before next iteration's writes
    }
}

extern "C" void kernel_launch(void* const* d_in, const int* in_sizes, int n_in,
                              void* d_out, int out_size, void* d_ws, size_t ws_size,
                              hipStream_t stream) {
    (void)in_sizes; (void)n_in; (void)out_size; (void)d_ws; (void)ws_size;
    const float* q = (const float*)d_in[0];
    const float* k = (const float*)d_in[1];
    const float* v = (const float*)d_in[2];
    const float* s = (const float*)d_in[3];
    float* out = (float*)d_out;

    // dynamic LDS: (2*128 + 256) f32 + (4*16384 + 2*4096 + 16384) f16 = 182272 bytes
    const size_t lds_bytes = (2 * 128 + 256) * sizeof(float) +
                             (4 * 128 * 128 + 2 * EW * 128 + 128 * 128) * sizeof(_Float16);
    dim3 grid(BB * HH * NSPL);   // 128 workgroups
    dim3 block(NTHR);            // 8 waves of 32
    lightning_attn_kernel<<<grid, block, lds_bytes, stream>>>(q, k, v, s, out);
}
